// TransformerBlock_17428977287510
// MI455X (gfx1250) — matrix operand order repaired and checked by reading, not compiled, not run on hardware
//
#include <hip/hip_runtime.h>
#include <hip/hip_bf16.h>

typedef _Float16 f16;
typedef _Float16 v16h __attribute__((ext_vector_type(16)));
typedef _Float16 v8h  __attribute__((ext_vector_type(8)));
typedef float    v8f  __attribute__((ext_vector_type(8)));

#define DEV static __device__ __forceinline__

constexpr int   Bb = 16, Ss = 2048, Dd = 1024;
constexpr int   BS = Bb * Ss;
constexpr float LN_EPS = 1e-5f;
// scale = 1/(sqrt(1024)+1e-8); folded with log2(e) so softmax runs on exp2 (v_exp_f32)
constexpr float SCALE_L2E = 1.4426950408889634f / 32.00000001f;

DEV v8f wmma_f16(v16h a, v16h b, v8f c) {
  return __builtin_amdgcn_wmma_f32_16x16x32_f16(false, a, false, b, (short)0, c, false, false);
}

DEV v8h ld8(const f16* p) {
  return *(const v8h*)__builtin_assume_aligned(p, 16);
}

// A-matrix 16x32 f16 fragment: lane(<16) row=lane%16; elems 0-7 <- k = 8*hi+0..7,
// elems 8-15 <- k = 16+8*hi+0..7   (ISA 7.12.2, 16-bit A 16x32)
DEV v16h ld_frag_a(const f16* row, int hi) {
  v8h x = ld8(row + 8 * hi);
  v8h y = ld8(row + 16 + 8 * hi);
  v16h r;
#pragma unroll
  for (int j = 0; j < 8; ++j) { r[j] = x[j]; r[j + 8] = y[j]; }
  return r;
}

// B-matrix 32x16 f16 fragment: lane holds column n=lane%16; elems 0-7 <- k = 8*hi+0..7,
// elems 8-15 <- k = 16+8*hi+0..7 (same order as A)
DEV v16h ld_frag_b(const f16* col, int hi) {
  v8h x = ld8(col + 8 * hi);
  v8h y = ld8(col + 16 + 8 * hi);
  v16h r;
#pragma unroll
  for (int j = 0; j < 8; ++j) { r[j] = x[j]; r[j + 8] = y[j]; }
  return r;
}

// ---------------------------------------------------------------------------
// Kernel 1: flash attention (32 queries/WG, 8 waves) + residual + LayerNorm1
// ---------------------------------------------------------------------------
constexpr int KSTR = 1040;  // halves per staged 1024-wide row (pad: align + banks)
constexpr int VSTR = 40;    // halves per V^T row (32 keys + pad)
constexpr int SSTR = 36;    // floats per partial-score row (32 + pad)
constexpr int PSTR = 40;    // halves per P row
constexpr int XSTR = 1028;  // floats per X row

constexpr int OFF_Q   = 0;
constexpr int OFF_K   = OFF_Q + 32 * KSTR * 2;          // 66560
constexpr int OFF_V   = OFF_K + 32 * KSTR * 2;          // 133120
constexpr int OFF_S   = OFF_V + 1024 * VSTR * 2;        // 215040
constexpr int OFF_P   = OFF_S + 8 * 32 * SSTR * 4;      // 251904
constexpr int OFF_AL  = OFF_P + 32 * PSTR * 2;          // 254464
constexpr int OFF_L   = OFF_AL + 32 * 4;
constexpr int OFF_RED = OFF_L + 32 * 4;
constexpr int OFF_MV  = OFF_RED + 32 * 8 * 2 * 4;
constexpr int SMEM1   = OFF_MV + 64 * 4;                // 257024 B (<320KB WGP LDS)
// X (32*XSTR*4 = 131584 B) overlays the dead Q+K staging region after the k-loop.

__global__ __launch_bounds__(256)
void attn_ln1_kernel(const float* __restrict__ Qg, const float* __restrict__ Kg,
                     const float* __restrict__ Vg, const float* __restrict__ gamma,
                     const float* __restrict__ beta, float* __restrict__ Xout) {
  extern __shared__ __align__(16) char smem[];
  f16*   Ql = (f16*)(smem + OFF_Q);
  f16*   Kl = (f16*)(smem + OFF_K);
  f16*   Vt = (f16*)(smem + OFF_V);
  float* Sp = (float*)(smem + OFF_S);
  f16*   Pl = (f16*)(smem + OFF_P);
  float* Al = (float*)(smem + OFF_AL);
  float* Ll = (float*)(smem + OFF_L);
  float* Rd = (float*)(smem + OFF_RED);
  float* Mv = (float*)(smem + OFF_MV);
  float* Xl = (float*)smem;

  const int t    = threadIdx.x;
  const int lane = t & 31, w = t >> 5, hi = lane >> 4, lo = lane & 15;
  const int bi   = blockIdx.x >> 6;
  const int q0   = (blockIdx.x & 63) * 32;
  const size_t base = (size_t)bi * Ss * Dd;

  // ---- stage Q tile 32x1024 fp32 -> f16 LDS ----
  {
    const int r = t >> 3, seg = (t & 7) * 128;
    const float* src = Qg + base + (size_t)(q0 + r) * Dd + seg;
    f16* dst = Ql + r * KSTR + seg;
#pragma unroll 8
    for (int i = 0; i < 128; i += 4) {
      float4 v = *(const float4*)(src + i);
      dst[i] = (f16)v.x; dst[i+1] = (f16)v.y; dst[i+2] = (f16)v.z; dst[i+3] = (f16)v.w;
    }
  }

  v8f o[2][8];
  const v8f vzero = {0.f,0.f,0.f,0.f,0.f,0.f,0.f,0.f};
#pragma unroll
  for (int mt = 0; mt < 2; ++mt)
#pragma unroll
    for (int nt = 0; nt < 8; ++nt) o[mt][nt] = vzero;

  float m_i = -3.0e38f, l_i = 0.0f;  // online-softmax state, wave 0: row == lane

  for (int kt = 0; kt < 64; ++kt) {
    const int k0 = kt * 32;
    // ---- cooperative stage: K row-major f16, V transposed f16 ----
    {
      const int r = t >> 3, seg = (t & 7) * 128;
      const float* ks = Kg + base + (size_t)(k0 + r) * Dd + seg;
      f16* kd = Kl + r * KSTR + seg;
#pragma unroll 8
      for (int i = 0; i < 128; i += 4) {
        float4 v = *(const float4*)(ks + i);
        kd[i] = (f16)v.x; kd[i+1] = (f16)v.y; kd[i+2] = (f16)v.z; kd[i+3] = (f16)v.w;
      }
      const float* vs = Vg + base + (size_t)(k0 + r) * Dd + seg;
#pragma unroll 8
      for (int i = 0; i < 128; i += 4) {
        float4 v = *(const float4*)(vs + i);
        Vt[(seg+i  ) * VSTR + r] = (f16)v.x;
        Vt[(seg+i+1) * VSTR + r] = (f16)v.y;
        Vt[(seg+i+2) * VSTR + r] = (f16)v.z;
        Vt[(seg+i+3) * VSTR + r] = (f16)v.w;
      }
    }
    __syncthreads();

    // ---- QK^T: each wave reduces its 128-dim slice (no redundant flops) ----
    v8f s[2][2];
    s[0][0] = vzero; s[0][1] = vzero; s[1][0] = vzero; s[1][1] = vzero;
#pragma unroll
    for (int kc = 0; kc < 4; ++kc) {
      const int dof = w * 128 + kc * 32;
      v16h a0  = ld_frag_a(Ql + (lo      ) * KSTR + dof, hi);
      v16h a1  = ld_frag_a(Ql + (16 + lo ) * KSTR + dof, hi);
      v16h kb0 = ld_frag_b(Kl + (lo      ) * KSTR + dof, hi);
      v16h kb1 = ld_frag_b(Kl + (16 + lo ) * KSTR + dof, hi);
      s[0][0] = wmma_f16(a0, kb0, s[0][0]);
      s[0][1] = wmma_f16(a0, kb1, s[0][1]);
      s[1][0] = wmma_f16(a1, kb0, s[1][0]);
      s[1][1] = wmma_f16(a1, kb1, s[1][1]);
    }
#pragma unroll
    for (int mt = 0; mt < 2; ++mt)
#pragma unroll
      for (int nt = 0; nt < 2; ++nt)
#pragma unroll
        for (int r = 0; r < 8; ++r)
          Sp[(w * 32 + mt * 16 + r + 8 * hi) * SSTR + nt * 16 + lo] = s[mt][nt][r];
    __syncthreads();

    // ---- reduce partials + online softmax (wave 0, one lane per query row) ----
    if (t < 32) {
      float sr[32];
#pragma unroll
      for (int j = 0; j < 32; ++j) sr[j] = 0.0f;
      for (int ww = 0; ww < 8; ++ww) {
        const float* sp = Sp + (ww * 32 + t) * SSTR;
#pragma unroll
        for (int j = 0; j < 32; ++j) sr[j] += sp[j];
      }
      float mx = m_i;
#pragma unroll
      for (int j = 0; j < 32; ++j) { sr[j] *= SCALE_L2E; mx = fmaxf(mx, sr[j]); }
      const float al = exp2f(m_i - mx);
      float ls = l_i * al;
#pragma unroll
      for (int j = 0; j < 32; ++j) {
        const float p = exp2f(sr[j] - mx);
        ls += p;
        Pl[t * PSTR + j] = (f16)p;
      }
      m_i = mx; l_i = ls;
      Al[t] = al; Ll[t] = ls;
    }
    __syncthreads();

    // ---- rescale O, accumulate P @ V over this wave's 128-dim slice ----
    float ar[2][8];
#pragma unroll
    for (int mt = 0; mt < 2; ++mt)
#pragma unroll
      for (int r = 0; r < 8; ++r) ar[mt][r] = Al[mt * 16 + r + 8 * hi];
    v16h p0 = ld_frag_a(Pl + (lo     ) * PSTR, hi);
    v16h p1 = ld_frag_a(Pl + (16 + lo) * PSTR, hi);
#pragma unroll
    for (int nt = 0; nt < 8; ++nt) {
      v16h vb = ld_frag_b(Vt + (w * 128 + nt * 16 + lo) * VSTR, hi);
#pragma unroll
      for (int r = 0; r < 8; ++r) { o[0][nt][r] *= ar[0][r]; o[1][nt][r] *= ar[1][r]; }
      o[0][nt] = wmma_f16(p0, vb, o[0][nt]);
      o[1][nt] = wmma_f16(p1, vb, o[1][nt]);
    }
    __syncthreads();
  }

  // ---- O / l -> X staging (overlays dead Q/K LDS) ----
  float il[2][8];
#pragma unroll
  for (int mt = 0; mt < 2; ++mt)
#pragma unroll
    for (int r = 0; r < 8; ++r) il[mt][r] = 1.0f / Ll[mt * 16 + r + 8 * hi];
#pragma unroll
  for (int mt = 0; mt < 2; ++mt)
#pragma unroll
    for (int nt = 0; nt < 8; ++nt)
#pragma unroll
      for (int r = 0; r < 8; ++r)
        Xl[(mt * 16 + r + 8 * hi) * XSTR + w * 128 + nt * 16 + lo] = o[mt][nt][r] * il[mt][r];
  __syncthreads();

  // ---- residual (+Q) and LayerNorm1 statistics ----
  {
    const int r = t >> 3, seg = (t & 7) * 128;
    const float* qs = Qg + base + (size_t)(q0 + r) * Dd + seg;
    float* xr = Xl + r * XSTR + seg;
    float sum = 0.f, sq = 0.f;
#pragma unroll 8
    for (int i = 0; i < 128; i += 4) {
      float4 q = *(const float4*)(qs + i);
      float x0 = xr[i] + q.x, x1 = xr[i+1] + q.y, x2 = xr[i+2] + q.z, x3 = xr[i+3] + q.w;
      xr[i] = x0; xr[i+1] = x1; xr[i+2] = x2; xr[i+3] = x3;
      sum += x0 + x1 + x2 + x3;
      sq  += x0*x0 + x1*x1 + x2*x2 + x3*x3;
    }
    Rd[(r * 8 + (t & 7)) * 2 + 0] = sum;
    Rd[(r * 8 + (t & 7)) * 2 + 1] = sq;
  }
  __syncthreads();
  if (t < 32) {
    float sum = 0.f, sq = 0.f;
#pragma unroll
    for (int p = 0; p < 8; ++p) { sum += Rd[(t * 8 + p) * 2]; sq += Rd[(t * 8 + p) * 2 + 1]; }
    const float mean = sum * (1.0f / Dd);
    const float var  = sq * (1.0f / Dd) - mean * mean;
    Mv[t] = mean; Mv[32 + t] = rsqrtf(var + LN_EPS);
  }
  __syncthreads();
  {
    const int r = t >> 3, seg = (t & 7) * 128;
    const float mean = Mv[r], rstd = Mv[32 + r];
    const float* xr = Xl + r * XSTR + seg;
    float* dst = Xout + base + (size_t)(q0 + r) * Dd + seg;
#pragma unroll 8
    for (int i = 0; i < 128; i += 4) {
      float4 g  = *(const float4*)(gamma + seg + i);
      float4 bt = *(const float4*)(beta + seg + i);
      float4 y;
      y.x = (xr[i]   - mean) * rstd * g.x + bt.x;
      y.y = (xr[i+1] - mean) * rstd * g.y + bt.y;
      y.z = (xr[i+2] - mean) * rstd * g.z + bt.z;
      y.w = (xr[i+3] - mean) * rstd * g.w + bt.w;
      *(float4*)(dst + i) = y;
    }
  }
}

// ---------------------------------------------------------------------------
// Kernel 2: H = relu(X @ W1^T + b1) -> f16 ws.  64x128 tile per WG, 8 waves.
// ---------------------------------------------------------------------------
__global__ __launch_bounds__(256)
void ffn1_kernel(const float* __restrict__ X, const float* __restrict__ W1,
                 const float* __restrict__ b1, f16* __restrict__ H) {
  __shared__ __align__(16) f16 Xa[64 * 72];
  __shared__ __align__(16) f16 Wb[128 * 72];
  const int t = threadIdx.x, lane = t & 31, w = t >> 5, hi = lane >> 4, lo = lane & 15;
  const int m0 = blockIdx.x * 64, n0 = blockIdx.y * 128;
  const int rb = (w & 3) * 16, cb = (w >> 2) * 64;
  const v8f vzero = {0.f,0.f,0.f,0.f,0.f,0.f,0.f,0.f};
  v8f acc[4] = {vzero, vzero, vzero, vzero};

  for (int k0 = 0; k0 < Dd; k0 += 64) {
    { // stage X chunk (fp32 -> f16)
      const int r = t >> 2, c = (t & 3) * 16;
      const float* src = X + (size_t)(m0 + r) * Dd + k0 + c;
      f16* dst = Xa + r * 72 + c;
#pragma unroll
      for (int i = 0; i < 16; i += 4) {
        float4 v = *(const float4*)(src + i);
        dst[i] = (f16)v.x; dst[i+1] = (f16)v.y; dst[i+2] = (f16)v.z; dst[i+3] = (f16)v.w;
      }
    }
    { // stage W1 chunk (fp32 -> f16)
      const int r = t >> 1, c = (t & 1) * 32;
      const float* src = W1 + (size_t)(n0 + r) * Dd + k0 + c;
      f16* dst = Wb + r * 72 + c;
#pragma unroll
      for (int i = 0; i < 32; i += 4) {
        float4 v = *(const float4*)(src + i);
        dst[i] = (f16)v.x; dst[i+1] = (f16)v.y; dst[i+2] = (f16)v.z; dst[i+3] = (f16)v.w;
      }
    }
    __syncthreads();
#pragma unroll
    for (int kc = 0; kc < 64; kc += 32) {
      v16h a = ld_frag_a(Xa + (rb + lo) * 72 + kc, hi);
#pragma unroll
      for (int nt = 0; nt < 4; ++nt) {
        v16h bm = ld_frag_b(Wb + (cb + nt * 16 + lo) * 72 + kc, hi);
        acc[nt] = wmma_f16(a, bm, acc[nt]);
      }
    }
    __syncthreads();
  }
#pragma unroll
  for (int nt = 0; nt < 4; ++nt) {
    const int n = n0 + cb + nt * 16 + lo;
    const float bias = b1[n];
#pragma unroll
    for (int r = 0; r < 8; ++r) {
      const int m = m0 + rb + r + 8 * hi;
      H[(size_t)m * Dd + n] = (f16)fmaxf(acc[nt][r] + bias, 0.0f);
    }
  }
}

// ---------------------------------------------------------------------------
// Kernel 3: F = H @ W2^T + b2 -> d_out (fp32).  Same tiling, f16 A input.
// ---------------------------------------------------------------------------
__global__ __launch_bounds__(256)
void ffn2_kernel(const f16* __restrict__ H, const float* __restrict__ W2,
                 const float* __restrict__ b2, float* __restrict__ F) {
  __shared__ __align__(16) f16 Xa[64 * 72];
  __shared__ __align__(16) f16 Wb[128 * 72];
  const int t = threadIdx.x, lane = t & 31, w = t >> 5, hi = lane >> 4, lo = lane & 15;
  const int m0 = blockIdx.x * 64, n0 = blockIdx.y * 128;
  const int rb = (w & 3) * 16, cb = (w >> 2) * 64;
  const v8f vzero = {0.f,0.f,0.f,0.f,0.f,0.f,0.f,0.f};
  v8f acc[4] = {vzero, vzero, vzero, vzero};

  for (int k0 = 0; k0 < Dd; k0 += 64) {
    { // stage H chunk (already f16)
      const int r = t >> 2, c = (t & 3) * 16;
      const f16* src = H + (size_t)(m0 + r) * Dd + k0 + c;
      f16* dst = Xa + r * 72 + c;
      *(v8h*)dst       = *(const v8h*)__builtin_assume_aligned(src, 16);
      *(v8h*)(dst + 8) = *(const v8h*)__builtin_assume_aligned(src + 8, 16);
    }
    { // stage W2 chunk (fp32 -> f16)
      const int r = t >> 1, c = (t & 1) * 32;
      const float* src = W2 + (size_t)(n0 + r) * Dd + k0 + c;
      f16* dst = Wb + r * 72 + c;
#pragma unroll
      for (int i = 0; i < 32; i += 4) {
        float4 v = *(const float4*)(src + i);
        dst[i] = (f16)v.x; dst[i+1] = (f16)v.y; dst[i+2] = (f16)v.z; dst[i+3] = (f16)v.w;
      }
    }
    __syncthreads();
#pragma unroll
    for (int kc = 0; kc < 64; kc += 32) {
      v16h a = ld_frag_a(Xa + (rb + lo) * 72 + kc, hi);
#pragma unroll
      for (int nt = 0; nt < 4; ++nt) {
        v16h bm = ld_frag_b(Wb + (cb + nt * 16 + lo) * 72 + kc, hi);
        acc[nt] = wmma_f16(a, bm, acc[nt]);
      }
    }
    __syncthreads();
  }
#pragma unroll
  for (int nt = 0; nt < 4; ++nt) {
    const int n = n0 + cb + nt * 16 + lo;
    const float bias = b2[n];
#pragma unroll
    for (int r = 0; r < 8; ++r) {
      const int m = m0 + rb + r + 8 * hi;
      F[(size_t)m * Dd + n] = acc[nt][r] + bias;
    }
  }
}

// ---------------------------------------------------------------------------
// Kernel 4: out = LayerNorm(F + X). One wave32 per row; shuffle reductions.
// ---------------------------------------------------------------------------
__global__ __launch_bounds__(256)
void ln2_kernel(const float* __restrict__ X, const float* __restrict__ gamma,
                const float* __restrict__ beta, float* __restrict__ Out) {
  const int lane = threadIdx.x & 31, w = threadIdx.x >> 5;
  const size_t row = (size_t)blockIdx.x * 8 + w;
  float* orow = Out + row * Dd;
  const float* xrow = X + row * Dd;
  float v[32];
  float sum = 0.f, sq = 0.f;
#pragma unroll
  for (int i = 0; i < 8; ++i) {
    const int c = i * 128 + lane * 4;
    float4 f = *(const float4*)(orow + c);
    float4 x = *(const float4*)(xrow + c);
    v[4*i+0] = f.x + x.x; v[4*i+1] = f.y + x.y; v[4*i+2] = f.z + x.z; v[4*i+3] = f.w + x.w;
    sum += v[4*i] + v[4*i+1] + v[4*i+2] + v[4*i+3];
    sq  += v[4*i]*v[4*i] + v[4*i+1]*v[4*i+1] + v[4*i+2]*v[4*i+2] + v[4*i+3]*v[4*i+3];
  }
#pragma unroll
  for (int off = 16; off > 0; off >>= 1) {
    sum += __shfl_xor(sum, off, 32);
    sq  += __shfl_xor(sq,  off, 32);
  }
  const float mean = sum * (1.0f / Dd);
  const float rstd = rsqrtf(sq * (1.0f / Dd) - mean * mean + LN_EPS);
#pragma unroll
  for (int i = 0; i < 8; ++i) {
    const int c = i * 128 + lane * 4;
    float4 g  = *(const float4*)(gamma + c);
    float4 bt = *(const float4*)(beta + c);
    float4 y;
    y.x = (v[4*i+0] - mean) * rstd * g.x + bt.x;
    y.y = (v[4*i+1] - mean) * rstd * g.y + bt.y;
    y.z = (v[4*i+2] - mean) * rstd * g.z + bt.z;
    y.w = (v[4*i+3] - mean) * rstd * g.w + bt.w;
    *(float4*)(orow + c) = y;
  }
}

// ---------------------------------------------------------------------------
extern "C" void kernel_launch(void* const* d_in, const int* in_sizes, int n_in,
                              void* d_out, int out_size, void* d_ws, size_t ws_size,
                              hipStream_t stream) {
  (void)in_sizes; (void)n_in; (void)out_size; (void)ws_size;
  const float* Q  = (const float*)d_in[0];
  const float* K  = (const float*)d_in[1];
  const float* V  = (const float*)d_in[2];
  const float* W1 = (const float*)d_in[3];
  const float* b1 = (const float*)d_in[4];
  const float* W2 = (const float*)d_in[5];
  const float* b2 = (const float*)d_in[6];
  const float* g  = (const float*)d_in[7];
  const float* be = (const float*)d_in[8];
  float* out = (float*)d_out;

  // Workspace: X fp32 (128 MB) then H f16 (64 MB). Fully rewritten every call.
  float* Xws = (float*)d_ws;
  f16*   Hws = (f16*)((char*)d_ws + (size_t)BS * Dd * sizeof(float));

  attn_ln1_kernel<<<dim3(Bb * (Ss / 32)), dim3(256), SMEM1, stream>>>(Q, K, V, g, be, Xws);
  ffn1_kernel<<<dim3(BS / 64, Dd / 128), dim3(256), 0, stream>>>(Xws, W1, b1, Hws);
  ffn2_kernel<<<dim3(BS / 64, Dd / 128), dim3(256), 0, stream>>>(Hws, W2, b2, out);
  ln2_kernel<<<dim3(BS / 8), dim3(256), 0, stream>>>(Xws, g, be, out);
}
